// MimoLTI_21225728377309
// MI455X (gfx1250) — compile-verified, matrix operand order checked by reading, not run on hardware
//
#include <hip/hip_runtime.h>

typedef __attribute__((ext_vector_type(16))) _Float16 v16h;
typedef __attribute__((ext_vector_type(8)))  float    v8f;
typedef __attribute__((ext_vector_type(4)))  unsigned u32x4;
typedef __attribute__((ext_vector_type(8)))  unsigned u32x8;

#define T_LEN  16384
#define O_CH   64
#define I_CH   64
#define NB     16
#define NA     15
#define BLK    16                     // timesteps per WMMA tile
#define UNROLL 4                      // tiles per barrier phase (64 steps)
#define TPB    128                    // 4 waves, each owns 16 o-columns
#define CHUNK  1024                   // timesteps emitted per workgroup
#define WARM   128                    // warm-up samples (state influence <1e-10)
#define NCHUNK (T_LEN / CHUNK)        // 16 time chunks
#define U_TOT  (NB + WARM + CHUNK)    // staged u samples per WG
#define PHASES ((WARM + CHUNK) / (UNROLL * BLK))  // 18
#define WARMPH (WARM / (UNROLL * BLK))            // 2 silent phases

__global__ void zero_out_kernel(float* __restrict__ out, int n) {
  int idx = blockIdx.x * blockDim.x + threadIdx.x;
  if (idx < n) out[idx] = 0.0f;
}

// TDM 2D tile load (Global -> LDS), 4-byte elements.
// tile_x elems/row, tile_y rows, row stride = stride_elems; rows land
// contiguously in LDS at lds_byte_addr. 2-group D# (<=2D tensor).
__device__ __forceinline__ void tdm_load_2d_f32(unsigned lds_byte_addr,
                                                const void* gaddr,
                                                unsigned tile_x, unsigned tile_y,
                                                unsigned stride_elems) {
  unsigned long long ga = (unsigned long long)gaddr;
  u32x4 g0;
  g0[0] = 1u;                                   // count=1 (valid), user mode
  g0[1] = lds_byte_addr;                        // D#.lds_addr
  g0[2] = (unsigned)(ga & 0xffffffffu);         // global_addr[31:0]
  g0[3] = (unsigned)((ga >> 32) & 0x01ffffffu)  // global_addr[56:32]
          | (2u << 30);                         // type=2 ("image")
  u32x8 g1;
  g1[0] = (2u << 16);                           // wg_mask=0, data_size=4B
  g1[1] = (tile_x & 0xffffu) << 16;             // tensor_dim0[15:0]
  g1[2] = (tile_y & 0xffffu) << 16;             // dim0[31:16]=0, tensor_dim1 lo
  g1[3] = (tile_x & 0xffffu) << 16;             // dim1[31:16]=0, tile_dim0
  g1[4] = (tile_y & 0xffffu);                   // tile_dim1, tile_dim2=0
  g1[5] = stride_elems;                         // tensor_dim0_stride[31:0]
  g1[6] = 0u;
  g1[7] = 0u;
  asm volatile("tensor_load_to_lds %0, %1" :: "s"(g0), "s"(g1) : "memory");
}

__global__ __launch_bounds__(TPB)
void mimo_iir_kernel(const float* __restrict__ u,     // (T, I)
                     const float* __restrict__ bco,   // (O, I, NB)
                     const float* __restrict__ aco,   // (O, I, NA)
                     float* __restrict__ out) {       // (T, O), pre-zeroed
  const int i      = blockIdx.x;                 // input channel
  const int c      = blockIdx.y;                 // time chunk
  const int tid    = threadIdx.x;
  const int lane   = tid & 31;
  const int wave   = tid >> 5;
  const int t_base = c * CHUNK - WARM;           // first staged timestep (may be <0)

  __shared__ __align__(16) _Float16 u_sh[U_TOT];              // 2.3 KB
  __shared__ __align__(16) float    v_sh[UNROLL * BLK * O_CH]; // 16 KB
  __shared__ __align__(16) float    b_sh[O_CH * NB];          // 4 KB
  __shared__ __align__(16) float    a_sh[O_CH * NA];          // 3.75 KB

  // ---- TDM: strided coefficient tiles for channel i -> LDS (wave 0 issues).
  if (tid < 32) {
    tdm_load_2d_f32((unsigned)(unsigned long long)(&b_sh[0]),
                    bco + (size_t)i * NB, NB, O_CH, I_CH * NB);
    tdm_load_2d_f32((unsigned)(unsigned long long)(&a_sh[0]),
                    aco + (size_t)i * NA, NA, O_CH, I_CH * NA);
    __builtin_amdgcn_s_wait_tensorcnt(0);
  }

  // ---- Stage this chunk's u slice (f16); t<0 -> 0 (true zero init state).
  for (int j = tid; j < U_TOT; j += TPB) {
    const int t = t_base - NB + j;
    u_sh[j] = (t >= 0) ? (_Float16)u[t * I_CH + i] : (_Float16)0.0f;
  }
  __syncthreads();

  // ---- Constant B fragment: wave w covers o in [16w, 16w+16).
  // 16-bit B 32x16 layout (mirror of A): lane<16 -> N=lane, K=0..7 in v0..3,
  // K=16..23 in v4..7; lane>=16 -> N=lane-16, K=8..15 / 24..31. K>=16 zeroed.
  const int row    = lane & 15;
  const int hi     = lane >> 4;
  const int o_base = wave * 16;
  v16h bfrag;
#pragma unroll
  for (int e = 0; e < 8; ++e) {
    const int k = e + 8 * hi;
    bfrag[e]     = (_Float16)b_sh[(o_base + row) * NB + k];
    bfrag[e + 8] = (_Float16)0.0f;
  }

  // ---- Recurrence state: lane tid<64 owns filter (o=tid, i).
  float yst[NA];
  float ac[NA];
  if (tid < O_CH) {
#pragma unroll
    for (int k = 0; k < NA; ++k) {
      yst[k] = 0.0f;
      ac[k]  = a_sh[tid * NA + k];
    }
  }
  __syncthreads();

  const float inv_i = 1.0f / (float)I_CH;

  for (int ph = 0; ph < PHASES; ++ph) {
    // ---- FIR: 4 16x16 tiles via WMMA. V[m][o] = sum_k b[o,i,k]*u[t-k,i].
    // A = 16x16 Toeplitz of u (zero-padded to K=32), shared across all 64 o.
#pragma unroll
    for (int blk = 0; blk < UNROLL; ++blk) {
      const int tloc = ph * (UNROLL * BLK) + blk * BLK;  // local tile time
      v16h afrag;
#pragma unroll
      for (int e = 0; e < 8; ++e) {
        const int k = e + 8 * hi;
        afrag[e]     = u_sh[NB + tloc + row - k];
        afrag[e + 8] = (_Float16)0.0f;
      }
      v8f cacc = {};
      cacc = __builtin_amdgcn_wmma_f32_16x16x32_f16(
          /*neg_a=*/false, afrag, /*neg_b=*/false, bfrag,
          /*c_mod=*/(short)0, cacc, /*reuse_a=*/false, /*reuse_b=*/false);
      // C 16x16 f32 layout: vgpr r -> M = r + 8*hi, N = lane&15.
#pragma unroll
      for (int r = 0; r < 8; ++r) {
        const int m = r + 8 * hi;
        v_sh[(blk * BLK + m) * O_CH + o_base + row] = cacc[r];
      }
    }
    __syncthreads();

    // ---- Feedback recurrence: y[t] = v[t] - sum_k a_k*y[t-k], lane = o.
    if (tid < O_CH) {
      const int  o    = tid;
      const bool emit = (ph >= WARMPH);   // silent during warm-up
#pragma unroll
      for (int m = 0; m < UNROLL * BLK; ++m) {
        float y = v_sh[m * O_CH + o];
#pragma unroll
        for (int k = 0; k < NA; ++k) y -= ac[k] * yst[k];
#pragma unroll
        for (int k = NA - 1; k > 0; --k) yst[k] = yst[k - 1];
        yst[0] = y;
        if (emit) {
          const int t = t_base + ph * (UNROLL * BLK) + m;  // >= c*CHUNK >= 0
          // mean over i: accumulate y/64 into out[t, o] across the 64 i-WGs.
          __hip_atomic_fetch_add(&out[t * O_CH + o], y * inv_i,
                                 __ATOMIC_RELAXED, __HIP_MEMORY_SCOPE_AGENT);
        }
      }
    }
    __syncthreads();  // protect v_sh before next phase overwrites it
  }
}

extern "C" void kernel_launch(void* const* d_in, const int* in_sizes, int n_in,
                              void* d_out, int out_size, void* d_ws, size_t ws_size,
                              hipStream_t stream) {
  const float* u   = (const float*)d_in[0];  // (T, I)
  const float* bco = (const float*)d_in[1];  // (O, I, NB)
  const float* aco = (const float*)d_in[2];  // (O, I, NA)
  float* out = (float*)d_out;                // (T, O)

  const int n = out_size;  // T*O
  zero_out_kernel<<<(n + 255) / 256, 256, 0, stream>>>(out, n);
  dim3 grid(I_CH, NCHUNK);  // 64 channels x 16 time chunks = 1024 WGs
  mimo_iir_kernel<<<grid, TPB, 0, stream>>>(u, bco, aco, out);
}
